// GraphTransformer_6794638262757
// MI455X (gfx1250) — matrix operand order repaired and checked by reading, not compiled, not run on hardware
//
#include <hip/hip_runtime.h>

// ---------------------------------------------------------------------------
// Problem constants (from reference)
// ---------------------------------------------------------------------------
constexpr int B    = 32;
constexpr int NPG  = 256;
constexpr int EG   = 2048;
constexpr int F    = 128;
constexpr int EDIM = 16;
constexpr int ESM  = 1280;
constexpr int NN   = B * NPG;     // 8192 nodes
constexpr int E    = B * EG;      // 65536 edges
constexpr int H    = 4;
constexpr int Dh   = 128;
constexpr int HD   = H * Dh;      // 512
constexpr int GW   = 2 * F + ESM; // 1536
constexpr float ATTN_SCALE = 0.08838834764831845f;  // 1/sqrt(128)

typedef __attribute__((ext_vector_type(16))) _Float16 v16h;
typedef __attribute__((ext_vector_type(8)))  _Float16 v8h;
typedef __attribute__((ext_vector_type(8)))  float    v8f;

// ---------------------------------------------------------------------------
// Precompute: f32 -> f16 with row padding; f32 [K,N] -> f16 [NPAD,KPAD] transpose
// (one-time per launch; weights are static, activations converted per pass)
// ---------------------------------------------------------------------------
__global__ void cvt_pad_f16(const float* __restrict__ src, _Float16* __restrict__ dst,
                            int M, int K, int KPAD)
{
    int i = blockIdx.x * 256 + threadIdx.x;
    if (i >= M * KPAD) return;
    int r = i / KPAD, c = i % KPAD;
    dst[i] = (_Float16)((c < K) ? src[(size_t)r * K + c] : 0.0f);
}

__global__ void transpose_f16(const float* __restrict__ W, _Float16* __restrict__ Wt,
                              int K, int N, int KPAD, int NPAD)
{
    int i = blockIdx.x * 256 + threadIdx.x;
    if (i >= NPAD * KPAD) return;
    int n = i / KPAD, k = i % KPAD;
    Wt[i] = (_Float16)((n < N && k < K) ? W[(size_t)k * N + n] : 0.0f);
}

// ---------------------------------------------------------------------------
// Templated WMMA GEMM:  C[M,N] = A16[M,K] x Wt16[N,K]^T  (all-f16 operands,
// f32 accumulate) with epilogue:
//   MODE 0: C = AB + bias             -> f32 out
//   MODE 1: C = AB                    -> f16 out
//   MODE 2: C = relu(AB + bias)       -> f16 out
//   MODE 3: C = BN(relu(AB + bias))   -> f32 out AND f16 out
// Block: 256 threads = 8 waves; wave tile 16x16; per K-chunk each lane issues
// 2x b128 A loads + 2x b128 B loads (exact WMMA fragment layouts, contiguous,
// 16B aligned) + one v_wmma_f32_16x16x32_f16. Wt16 is pre-transposed/padded so
// there are no runtime guards in the hot loop.
// ---------------------------------------------------------------------------
template<int K, int N, int MODE>
__global__ void wmma_gemm_t(const _Float16* __restrict__ A, const _Float16* __restrict__ Wt,
                            const float* __restrict__ bias,
                            float* __restrict__ outF, _Float16* __restrict__ outH,
                            const float* __restrict__ gm, const float* __restrict__ bt2,
                            const float* __restrict__ rm, const float* __restrict__ rv)
{
    static_assert(K % 32 == 0, "K must be a multiple of 32 (zero-padded)");
    constexpr int NPAD = (N + 15) & ~15;

    const int lane   = threadIdx.x & 31;
    const int wave   = threadIdx.x >> 5;
    const int tn     = blockIdx.x * 128 + wave * 16;
    if (tn >= NPAD) return;                       // wave-uniform (N=10 case)
    const int tm     = blockIdx.y * 16;
    const int laneHi = lane >> 4;
    const int lane15 = lane & 15;

    // A fragment: lane row = tm+lane15; halves at K offsets {+0..7, +16..23} (+8 if laneHi)
    const _Float16* ap = A  + (size_t)(tm + lane15) * K + laneHi * 8;
    // B fragment: lane col = tn+lane15; halves at K offsets {+0..7, +16..23} (+8 if laneHi)
    const _Float16* bp = Wt + (size_t)(tn + lane15) * K + laneHi * 8;

    v8f c = {};
#pragma unroll 4
    for (int ch = 0; ch < K / 32; ++ch) {
        const int kb = ch * 32;
        v8h a0 = *(const v8h*)(ap + kb);
        v8h a1 = *(const v8h*)(ap + kb + 16);
        v8h b0 = *(const v8h*)(bp + kb);
        v8h b1 = *(const v8h*)(bp + kb + 16);
        v16h af = __builtin_shufflevector(a0, a1, 0,1,2,3,4,5,6,7,8,9,10,11,12,13,14,15);
        v16h bf = __builtin_shufflevector(b0, b1, 0,1,2,3,4,5,6,7,8,9,10,11,12,13,14,15);
        c = __builtin_amdgcn_wmma_f32_16x16x32_f16(false, af, false, bf,
                                                   (short)0, c, false, false);
        if (ch + 1 < K / 32)
            __builtin_prefetch(ap + kb + 32, 0, 0);
    }

    const int col = tn + lane15;
#pragma unroll
    for (int r = 0; r < 8; ++r) {
        const int row = tm + r + laneHi * 8;      // D layout: VGPR r -> M=r / M=r+8
        if (N % 16 == 0 || col < N) {
            float val = c[r];
            if constexpr (MODE != 1) val += bias[col];
            if constexpr (MODE >= 2) val = fmaxf(val, 0.0f);
            if constexpr (MODE == 3)
                val = (val - rm[col]) / sqrtf(rv[col] + 1e-5f) * gm[col] + bt2[col];
            if constexpr (MODE == 0 || MODE == 3) outF[(size_t)row * N + col] = val;
            if constexpr (MODE != 0)              outH[(size_t)row * N + col] = (_Float16)val;
        }
    }
}

template<int K, int N, int MODE>
static void run_gemm(hipStream_t st, const _Float16* A, const _Float16* Wt,
                     const float* bias, float* outF, _Float16* outH,
                     const float* gm, const float* bt2,
                     const float* rm, const float* rv, int M)
{
    dim3 grid((unsigned)((N + 127) / 128), (unsigned)(M / 16), 1);
    wmma_gemm_t<K, N, MODE><<<grid, dim3(256), 0, st>>>(A, Wt, bias, outF, outH,
                                                        gm, bt2, rm, rv);
}

// ---------------------------------------------------------------------------
// Utility fills
// ---------------------------------------------------------------------------
__global__ void fill_f32(float* p, float v, int n) {
    int i = blockIdx.x * 256 + threadIdx.x;
    if (i < n) p[i] = v;
}
__global__ void fill_u8(unsigned char* p, unsigned char v, int n) {
    int i = blockIdx.x * 256 + threadIdx.x;
    if (i < n) p[i] = v;
}

__device__ inline void atomicMaxFloat(float* addr, float val) {
    unsigned int* ua  = (unsigned int*)addr;
    unsigned int  old = *ua;
    while (true) {
        float f = __uint_as_float(old);
        if (f >= val) break;
        unsigned int assumed = old;
        old = atomicCAS(ua, assumed, __float_as_uint(val));
        if (old == assumed) break;
    }
}

// ---------------------------------------------------------------------------
// Attention pass 1: per-edge scores a[e,h] = scale * q[dst]·(k[src]+e16), and
// CAS-max into per-(node,head) maxima. One wave per edge.
// ---------------------------------------------------------------------------
__global__ void attn_score(const float* __restrict__ q, const float* __restrict__ k,
                           const _Float16* __restrict__ e16,
                           const int* __restrict__ src, const int* __restrict__ dst,
                           const unsigned char* __restrict__ emask,
                           float* __restrict__ a, float* __restrict__ m)
{
    int eIdx = blockIdx.x * 8 + (threadIdx.x >> 5);
    int lane = threadIdx.x & 31;
    if (eIdx >= E || !emask[eIdx]) return;
    int s = src[eIdx], d = dst[eIdx];
    const float*    qr = q   + (size_t)d * HD;
    const float*    kr = k   + (size_t)s * HD;
    const _Float16* er = e16 + (size_t)eIdx * HD;
#pragma unroll
    for (int h = 0; h < H; ++h) {
        float acc = 0.0f;
        for (int dd = lane; dd < Dh; dd += 32) {
            int idx = h * Dh + dd;
            acc += qr[idx] * (kr[idx] + (float)er[idx]);
        }
#pragma unroll
        for (int off = 16; off; off >>= 1) acc += __shfl_xor(acc, off, 32);
        if (lane == 0) {
            float val = acc * ATTN_SCALE;
            a[(size_t)eIdx * H + h] = val;
            atomicMaxFloat(&m[(size_t)d * H + h], val);
        }
    }
}

// Pass 2: ea = exp(a - m[dst]) * emask; accumulate denominators.
__global__ void attn_softmax(const int* __restrict__ dst,
                             const unsigned char* __restrict__ emask,
                             float* __restrict__ a, const float* __restrict__ m,
                             float* __restrict__ den)
{
    int i = blockIdx.x * 256 + threadIdx.x;   // over E*H
    if (i >= E * H) return;
    int e = i / H, h = i % H;
    float ea = 0.0f;
    if (emask[e]) {
        float mm = fmaxf(m[(size_t)dst[e] * H + h], -1e30f);
        ea = __expf(a[i] - mm);
        atomicAdd(&den[(size_t)dst[e] * H + h], ea);
    }
    a[i] = ea;
}

// Pass 3: out[dst] += (v[src] + e16) * (ea / den[dst]). One wave per edge.
__global__ void attn_aggregate(const float* __restrict__ v, const _Float16* __restrict__ e16,
                               const int* __restrict__ src, const int* __restrict__ dst,
                               const unsigned char* __restrict__ emask,
                               const float* __restrict__ a, const float* __restrict__ den,
                               float* __restrict__ out)
{
    int eIdx = blockIdx.x * 8 + (threadIdx.x >> 5);
    int lane = threadIdx.x & 31;
    if (eIdx >= E || !emask[eIdx]) return;
    int s = src[eIdx], d = dst[eIdx];
    float attn[H];
#pragma unroll
    for (int h = 0; h < H; ++h)
        attn[h] = a[(size_t)eIdx * H + h] / fmaxf(den[(size_t)d * H + h], 1e-16f);
    for (int idx = lane; idx < HD; idx += 32) {
        int h = idx >> 7;                     // idx / Dh
        float msg = (v[(size_t)s * HD + idx] + (float)e16[(size_t)eIdx * HD + idx]) * attn[h];
        atomicAdd(&out[(size_t)d * HD + idx], msg);
    }
}

// ---------------------------------------------------------------------------
// Beta gate: bta = sigmoid([out, xr, out-xr] @ Wbeta);
// out16 <- f16( bta*xr + (1-bta)*out )   (direct f16 GEMM input)
// ---------------------------------------------------------------------------
__global__ void gate_beta(const float* __restrict__ out, const float* __restrict__ skip,
                          const float* __restrict__ Wb, _Float16* __restrict__ out16)
{
    int n    = blockIdx.x * 8 + (threadIdx.x >> 5);
    int lane = threadIdx.x & 31;
    if (n >= NN) return;
    float acc = 0.0f;
    for (int i = lane; i < HD; i += 32) {
        float o = out[(size_t)n * HD + i], xr = skip[(size_t)n * HD + i];
        acc += o * Wb[i] + xr * Wb[HD + i] + (o - xr) * Wb[2 * HD + i];
    }
#pragma unroll
    for (int off = 16; off; off >>= 1) acc += __shfl_xor(acc, off, 32);
    float bta = 1.0f / (1.0f + __expf(-acc));
    for (int i = lane; i < HD; i += 32) {
        float o = out[(size_t)n * HD + i], xr = skip[(size_t)n * HD + i];
        out16[(size_t)n * HD + i] = (_Float16)(bta * xr + (1.0f - bta) * o);
    }
}

// ---------------------------------------------------------------------------
// TopK pooling: score, rank-and-keep (scales f32 AND f16 h), emask, readout
// ---------------------------------------------------------------------------
__global__ void pool_score(const float* __restrict__ h, const float* __restrict__ w,
                           float* __restrict__ sc)
{
    int n    = blockIdx.x * 8 + (threadIdx.x >> 5);
    int lane = threadIdx.x & 31;
    if (n >= NN) return;
    float nw = 0.0f, acc = 0.0f;
    for (int i = lane; i < F; i += 32) {
        float wv = w[i];
        nw  += wv * wv;
        acc += h[(size_t)n * F + i] * wv;
    }
#pragma unroll
    for (int off = 16; off; off >>= 1) {
        nw  += __shfl_xor(nw, off, 32);
        acc += __shfl_xor(acc, off, 32);
    }
    if (lane == 0) sc[n] = tanhf(acc / sqrtf(nw));
}

// One block per graph (NPG threads). Stable descending rank == argsort(argsort(-s)).
__global__ void pool_rank(const float* __restrict__ sc, unsigned char* __restrict__ nmask,
                          float* __restrict__ h, _Float16* __restrict__ h16, int kkeep)
{
    __shared__ float ss[NPG];
    int b = blockIdx.x, t = threadIdx.x;
    int n = b * NPG + t;
    float sv = nmask[n] ? sc[n] : -3.0e38f;
    ss[t] = sv;
    __syncthreads();
    int rank = 0;
    for (int j = 0; j < NPG; ++j) {
        float o = ss[j];
        if (o > sv || (o == sv && j < t)) rank++;
    }
    bool keep = (rank < kkeep) && nmask[n];
    nmask[n] = keep ? 1 : 0;
    float mult = keep ? sc[n] : 0.0f;
    for (int f = 0; f < F; ++f) {
        size_t i = (size_t)n * F + f;
        float hv = h[i] * mult;
        h[i]   = hv;
        h16[i] = (_Float16)hv;
    }
}

__global__ void update_emask(const int* __restrict__ src, const int* __restrict__ dst,
                             const unsigned char* __restrict__ nmask,
                             unsigned char* __restrict__ emask)
{
    int e = blockIdx.x * 256 + threadIdx.x;
    if (e < E) emask[e] = (emask[e] && nmask[src[e]] && nmask[dst[e]]) ? 1 : 0;
}

// One block per graph, F threads (one per feature dim): [gmax | gmean].
__global__ void readout(const float* __restrict__ h, const unsigned char* __restrict__ nmask,
                        float* __restrict__ rep)
{
    int b = blockIdx.x, f = threadIdx.x;
    float mx = -1e30f, sm = 0.0f, cnt = 0.0f;
    for (int j = 0; j < NPG; ++j) {
        int n = b * NPG + j;
        bool kp = nmask[n] != 0;
        float hv = h[(size_t)n * F + f];
        if (kp) { mx = fmaxf(mx, hv); cnt += 1.0f; sm += hv; }
    }
    rep[(size_t)b * 2 * F + f]     = mx;
    rep[(size_t)b * 2 * F + F + f] = sm / fmaxf(cnt, 1.0f);
}

// g16 = f16([rep0 + rep1 , esm])  ->  [B, GW]
__global__ void build_g(const float* __restrict__ rep0, const float* __restrict__ rep1,
                        const float* __restrict__ esm, _Float16* __restrict__ g16)
{
    int i = blockIdx.x * 256 + threadIdx.x;
    if (i >= B * GW) return;
    int b = i / GW, c = i % GW;
    float v = (c < 2 * F) ? (rep0[(size_t)b * 2 * F + c] + rep1[(size_t)b * 2 * F + c])
                          : esm[(size_t)b * ESM + (c - 2 * F)];
    g16[i] = (_Float16)v;
}

// ---------------------------------------------------------------------------
// Host orchestration
// ---------------------------------------------------------------------------
extern "C" void kernel_launch(void* const* d_in, const int* in_sizes, int n_in,
                              void* d_out, int out_size, void* d_ws, size_t ws_size,
                              hipStream_t stream)
{
    (void)in_sizes; (void)n_in; (void)out_size; (void)ws_size;

    const float* x         = (const float*)d_in[0];
    const float* edge_attr = (const float*)d_in[1];
    const int*   eidx      = (const int*)  d_in[2];
    const int*   src       = eidx;
    const int*   dst       = eidx + E;
    const float* esm       = (const float*)d_in[4];
    const float* Wq    = (const float*)d_in[5];
    const float* bq    = (const float*)d_in[6];
    const float* Wk    = (const float*)d_in[7];
    const float* bk    = (const float*)d_in[8];
    const float* Wv    = (const float*)d_in[9];
    const float* bv    = (const float*)d_in[10];
    const float* We    = (const float*)d_in[11];
    const float* Wskip = (const float*)d_in[12];
    const float* bskip = (const float*)d_in[13];
    const float* Wbeta = (const float*)d_in[14];
    const float* Wt    = (const float*)d_in[15];
    const float* bt    = (const float*)d_in[16];
    const float* gamma = (const float*)d_in[17];
    const float* betab = (const float*)d_in[18];
    const float* rmean = (const float*)d_in[19];
    const float* rvar  = (const float*)d_in[20];
    const float* pw    = (const float*)d_in[21];
    const float* W1    = (const float*)d_in[22];
    const float* b1    = (const float*)d_in[23];
    const float* W2    = (const float*)d_in[24];
    const float* b2    = (const float*)d_in[25];
    const float* W3    = (const float*)d_in[26];
    const float* b3    = (const float*)d_in[27];

    // Workspace carve-out
    char*  ws  = (char*)d_ws;
    size_t off = 0;
    auto alloc = [&](size_t bytes) -> char* {
        char* p = ws + off;
        off += (bytes + 255) & ~(size_t)255;
        return p;
    };
    // f32 activation buffers
    float*     qb    = (float*)alloc((size_t)NN * HD * 4);
    float*     kb    = (float*)alloc((size_t)NN * HD * 4);
    float*     vb    = (float*)alloc((size_t)NN * HD * 4);
    float*     skipb = (float*)alloc((size_t)NN * HD * 4);
    float*     outb  = (float*)alloc((size_t)NN * HD * 4);
    float*     ab    = (float*)alloc((size_t)E * H * 4);
    float*     mb    = (float*)alloc((size_t)NN * H * 4);
    float*     denb  = (float*)alloc((size_t)NN * H * 4);
    float*     hA    = (float*)alloc((size_t)NN * F * 4);
    float*     hB    = (float*)alloc((size_t)NN * F * 4);
    float*     scb   = (float*)alloc((size_t)NN * 4);
    float*     rep0  = (float*)alloc((size_t)B * 2 * F * 4);
    float*     rep1  = (float*)alloc((size_t)B * 2 * F * 4);
    // f16 activation buffers
    _Float16*  e16    = (_Float16*)alloc((size_t)E * HD * 2);
    _Float16*  x16    = (_Float16*)alloc((size_t)NN * F * 2);
    _Float16*  ea16   = (_Float16*)alloc((size_t)E * 32 * 2);
    _Float16*  h16A   = (_Float16*)alloc((size_t)NN * F * 2);
    _Float16*  h16B   = (_Float16*)alloc((size_t)NN * F * 2);
    _Float16*  out16  = (_Float16*)alloc((size_t)NN * HD * 2);
    _Float16*  gcat16 = (_Float16*)alloc((size_t)B * GW * 2);
    _Float16*  m1_16  = (_Float16*)alloc((size_t)B * 512 * 2);
    _Float16*  m2_16  = (_Float16*)alloc((size_t)B * 256 * 2);
    // pre-transposed f16 weights  [NPAD][KPAD]
    _Float16*  wqT = (_Float16*)alloc((size_t)5 * HD * F * 2);
    _Float16*  wkT = (_Float16*)alloc((size_t)5 * HD * F * 2);
    _Float16*  wvT = (_Float16*)alloc((size_t)5 * HD * F * 2);
    _Float16*  wsT = (_Float16*)alloc((size_t)5 * HD * F * 2);
    _Float16*  weT = (_Float16*)alloc((size_t)5 * HD * 32 * 2);
    _Float16*  wtT = (_Float16*)alloc((size_t)5 * F * HD * 2);
    _Float16*  w1T = (_Float16*)alloc((size_t)512 * GW * 2);
    _Float16*  w2T = (_Float16*)alloc((size_t)256 * 512 * 2);
    _Float16*  w3T = (_Float16*)alloc((size_t)16 * 256 * 2);
    unsigned char* nmask = (unsigned char*)alloc(NN);
    unsigned char* emask = (unsigned char*)alloc(E);

    auto nb = [](size_t n) { return (unsigned)((n + 255) / 256); };

    // ---- one-time precompute: activations to f16, weights transposed f16 ----
    cvt_pad_f16<<<nb((size_t)NN * F), 256, 0, stream>>>(x, x16, NN, F, F);
    cvt_pad_f16<<<nb((size_t)E * 32), 256, 0, stream>>>(edge_attr, ea16, E, EDIM, 32);
    for (int li = 0; li < 5; ++li) {
        transpose_f16<<<nb((size_t)HD * F), 256, 0, stream>>>(
            Wq + (size_t)li * F * HD, wqT + (size_t)li * HD * F, F, HD, F, HD);
        transpose_f16<<<nb((size_t)HD * F), 256, 0, stream>>>(
            Wk + (size_t)li * F * HD, wkT + (size_t)li * HD * F, F, HD, F, HD);
        transpose_f16<<<nb((size_t)HD * F), 256, 0, stream>>>(
            Wv + (size_t)li * F * HD, wvT + (size_t)li * HD * F, F, HD, F, HD);
        transpose_f16<<<nb((size_t)HD * F), 256, 0, stream>>>(
            Wskip + (size_t)li * F * HD, wsT + (size_t)li * HD * F, F, HD, F, HD);
        transpose_f16<<<nb((size_t)HD * 32), 256, 0, stream>>>(
            We + (size_t)li * EDIM * HD, weT + (size_t)li * HD * 32, EDIM, HD, 32, HD);
        transpose_f16<<<nb((size_t)F * HD), 256, 0, stream>>>(
            Wt + (size_t)li * HD * F, wtT + (size_t)li * F * HD, HD, F, HD, F);
    }
    transpose_f16<<<nb((size_t)512 * GW), 256, 0, stream>>>(W1, w1T, GW, 512, GW, 512);
    transpose_f16<<<nb((size_t)256 * 512), 256, 0, stream>>>(W2, w2T, 512, 256, 512, 256);
    transpose_f16<<<nb((size_t)16 * 256), 256, 0, stream>>>(W3, w3T, 256, 10, 256, 16);

    fill_u8<<<nb(NN), 256, 0, stream>>>(nmask, 1, NN);
    fill_u8<<<nb(E), 256, 0, stream>>>(emask, 1, E);

    auto conv = [&](const _Float16* hin16, int li, float* hout, _Float16* hout16) {
        run_gemm<F, HD, 0>(stream, hin16, wqT + (size_t)li * HD * F, bq + (size_t)li * HD,
                           qb, nullptr, nullptr, nullptr, nullptr, nullptr, NN);
        run_gemm<F, HD, 0>(stream, hin16, wkT + (size_t)li * HD * F, bk + (size_t)li * HD,
                           kb, nullptr, nullptr, nullptr, nullptr, nullptr, NN);
        run_gemm<F, HD, 0>(stream, hin16, wvT + (size_t)li * HD * F, bv + (size_t)li * HD,
                           vb, nullptr, nullptr, nullptr, nullptr, nullptr, NN);
        run_gemm<F, HD, 0>(stream, hin16, wsT + (size_t)li * HD * F, bskip + (size_t)li * HD,
                           skipb, nullptr, nullptr, nullptr, nullptr, nullptr, NN);
        run_gemm<32, HD, 1>(stream, ea16, weT + (size_t)li * HD * 32, nullptr,
                            nullptr, e16, nullptr, nullptr, nullptr, nullptr, E);

        fill_f32<<<nb(NN * H), 256, 0, stream>>>(mb, -1e30f, NN * H);
        fill_f32<<<nb(NN * H), 256, 0, stream>>>(denb, 0.0f, NN * H);
        fill_f32<<<nb((size_t)NN * HD), 256, 0, stream>>>(outb, 0.0f, NN * HD);

        attn_score    <<<E / 8, 256, 0, stream>>>(qb, kb, e16, src, dst, emask, ab, mb);
        attn_softmax  <<<nb((size_t)E * H), 256, 0, stream>>>(dst, emask, ab, mb, denb);
        attn_aggregate<<<E / 8, 256, 0, stream>>>(vb, e16, src, dst, emask, ab, denb, outb);
        gate_beta     <<<NN / 8, 256, 0, stream>>>(outb, skipb, Wbeta + (size_t)li * 3 * HD,
                                                   out16);

        run_gemm<HD, F, 3>(stream, out16, wtT + (size_t)li * F * HD, bt + (size_t)li * F,
                           hout, hout16, gamma + (size_t)li * F, betab + (size_t)li * F,
                           rmean + (size_t)li * F, rvar + (size_t)li * F, NN);
    };

    conv(x16, 0, hA, h16A);
    float*    hcur  = hA;   float*    hnext  = hB;
    _Float16* h16c  = h16A; _Float16* h16n   = h16B;
    int p = 0;
    for (int i = 0; i < 4; ++i) {
        conv(h16c, i + 1, hnext, h16n);
        { float* t = hcur; hcur = hnext; hnext = t; }
        { _Float16* t = h16c; h16c = h16n; h16n = t; }
        if (i % 2 == 0) {
            int    kk  = (p == 0) ? (NPG / 2) : (NPG / 4);
            float* rep = (p == 0) ? rep0 : rep1;
            pool_score  <<<NN / 8, 256, 0, stream>>>(hcur, pw + (size_t)p * F, scb);
            pool_rank   <<<B, NPG, 0, stream>>>(scb, nmask, hcur, h16c, kk);
            update_emask<<<nb(E), 256, 0, stream>>>(src, dst, nmask, emask);
            readout     <<<B, F, 0, stream>>>(hcur, nmask, rep);
            p++;
        }
    }

    build_g<<<nb((size_t)B * GW), 256, 0, stream>>>(rep0, rep1, esm, gcat16);
    run_gemm<GW, 512, 2>(stream, gcat16, w1T, b1, nullptr, m1_16,
                         nullptr, nullptr, nullptr, nullptr, B);
    run_gemm<512, 256, 2>(stream, m1_16, w2T, b2, nullptr, m2_16,
                          nullptr, nullptr, nullptr, nullptr, B);
    run_gemm<256, 10, 0>(stream, m2_16, w3T, b3, (float*)d_out, nullptr,
                         nullptr, nullptr, nullptr, nullptr, B);
}
